// SigMax1_21852793602395
// MI455X (gfx1250) — matrix operand order repaired and checked by reading, not compiled, not run on hardware
//
#include <hip/hip_runtime.h>
#include <hip/hip_bf16.h>
#include <math.h>

// Problem geometry (from reference setup_inputs): B=8, T=4096, F=512
#define BDIM 8
#define TDIM 4096
#define FDIM 512
#define NFRAMES (BDIM * TDIM)          // 32768 rows
#define OUT_FEAT_ELEMS ((size_t)NFRAMES * FDIM)

typedef __attribute__((ext_vector_type(16))) _Float16 v16h;
typedef __attribute__((ext_vector_type(8)))  float    v8f;

// ---------------------------------------------------------------------------
// Phase 1: score[b,t] = sigmoid(dot(feat[b,t,:], W) + bias) via WMMA.
// One wave32 computes 16 frames. A-matrix = 16x32 f16 tile of feat,
// B-matrix = W chunk broadcast across all 16 columns, f32 accumulate.
//
// A layout (16-bit A 16x32, ISA 7.12.2): lane L<16 holds row M=L,
//   v16h[0..7] = K {k0..k0+7}, v16h[8..15] = K {k0+16..k0+23};
//   lane L>=16 holds row M=L-16, K {k0+8..k0+15} and {k0+24..k0+31}.
// B layout (32x16): lane n<16 = column n, v16h[0..7] = K {k0..k0+7}, v16h[8..15] = K {k0+16..k0+23};
//   lane n>=16 = column n-16, K {k0+8..k0+15} and {k0+24..k0+31}.
// D layout (16x16 f32): VGPR i, lanes 0-15: M=i; lanes 16-31: M=8+i.
// ---------------------------------------------------------------------------
__global__ __launch_bounds__(256)
void sigmax_score_wmma(const float* __restrict__ feat,
                       const float* __restrict__ W,
                       const float* __restrict__ bias,
                       float* __restrict__ score)   // [B*T]
{
    const int lane   = threadIdx.x & 31;
    const int waveId = (blockIdx.x * (blockDim.x >> 5)) + (threadIdx.x >> 5);
    const int frame0 = waveId * 16;                 // 16 frames per wave
    const int row    = frame0 + (lane & 15);
    const bool hiK   = (lane >= 16);
    const float* __restrict__ rowPtr = feat + (size_t)row * FDIM;

    v8f acc = {};
    for (int k0 = 0; k0 < FDIM; k0 += 32) {
        const int kbase = k0 + (hiK ? 8 : 0);
        v16h a;
#pragma unroll
        for (int i = 0; i < 8; ++i) {
            a[i]     = (_Float16)rowPtr[kbase + i];
            a[8 + i] = (_Float16)rowPtr[kbase + 16 + i];
        }
        const int wbase = k0 + (hiK ? 8 : 0);
        v16h bm;
#pragma unroll
        for (int i = 0; i < 8; ++i) {
            bm[i]     = (_Float16)W[wbase + i];
            bm[8 + i] = (_Float16)W[wbase + 16 + i];
        }
        // (neg_a, A, neg_b, B, c_mod, C, reuse_a, reuse_b)
        acc = __builtin_amdgcn_wmma_f32_16x16x32_f16(
            false, a, false, bm, (short)0, acc, false, false);
    }

    // Every D column holds the same dot products; lanes 0 and 16 extract them.
    const float bv = bias[0];
    if (lane == 0 || lane == 16) {
        const int mbase = (lane == 0) ? 0 : 8;
#pragma unroll
        for (int i = 0; i < 8; ++i) {
            float s = acc[i] + bv;
            s = 1.0f / (1.0f + __expf(-s));
            score[frame0 + mbase + i] = s;
        }
    }
}

// ---------------------------------------------------------------------------
// Phase 2: peak detect + stable compaction indices + hlens.
// One 1024-thread block per batch; 4 frames per thread; LDS inclusive scan.
// ---------------------------------------------------------------------------
__global__ __launch_bounds__(1024)
void sigmax_peaks_scan(const float* __restrict__ score,   // [B*T]
                       int* __restrict__ srcIdx,          // [B*T] ws
                       int* __restrict__ hlens_out)       // [B] (int bits in d_out tail)
{
    const int b   = blockIdx.x;           // 0..7
    const int tid = threadIdx.x;          // 0..1023
    const float* __restrict__ s = score + (size_t)b * TDIM;

    __shared__ int psum[1024];

    const int tbase = tid * 4;
    int flags[4];
    int cnt = 0;
#pragma unroll
    for (int i = 0; i < 4; ++i) {
        const int t = tbase + i;
        const float c = s[t];
        const float l = (t == 0)        ? c : s[t - 1];
        const float r = (t == TDIM - 1) ? c : s[t + 1];
        const int p = (c >= l && c >= r) ? 1 : 0;
        flags[i] = p;
        cnt += p;
    }
    psum[tid] = cnt;
    __syncthreads();

    // Hillis-Steele inclusive scan over 1024 partial counts.
    for (int off = 1; off < 1024; off <<= 1) {
        const int v   = psum[tid];
        const int add = (tid >= off) ? psum[tid - off] : 0;
        __syncthreads();
        psum[tid] = v + add;
        __syncthreads();
    }

    int pos = psum[tid] - cnt;            // exclusive prefix (stable order)
#pragma unroll
    for (int i = 0; i < 4; ++i) {
        if (flags[i]) srcIdx[(size_t)b * TDIM + (pos++)] = tbase + i;
    }
    if (tid == 0) hlens_out[b] = psum[1023];
}

// ---------------------------------------------------------------------------
// Phase 3: gather compacted rows / zero-fill tail. One block per output row,
// 128 threads x float4 = 512 floats. Pure streaming bandwidth.
// ---------------------------------------------------------------------------
__global__ __launch_bounds__(128)
void sigmax_gather(const float* __restrict__ feat,
                   const int* __restrict__ srcIdx,
                   const int* __restrict__ hlens,
                   float* __restrict__ out)
{
    const int rowId = blockIdx.x;         // 0..B*T-1
    const int b = rowId >> 12;            // /TDIM
    const int r = rowId & (TDIM - 1);
    float4* __restrict__ dst = (float4*)(out + (size_t)rowId * FDIM);
    const int tid = threadIdx.x;

    if (r < hlens[b]) {
        const int src = srcIdx[(size_t)b * TDIM + r];
        const float4* __restrict__ s4 =
            (const float4*)(feat + ((size_t)b * TDIM + src) * FDIM);
        dst[tid] = s4[tid];
    } else {
        dst[tid] = make_float4(0.f, 0.f, 0.f, 0.f);
    }
}

// ---------------------------------------------------------------------------
extern "C" void kernel_launch(void* const* d_in, const int* in_sizes, int n_in,
                              void* d_out, int out_size, void* d_ws, size_t ws_size,
                              hipStream_t stream) {
    const float* feat = (const float*)d_in[0];   // [B,T,F] f32
    const float* W    = (const float*)d_in[1];   // [F]
    const float* bias = (const float*)d_in[2];   // [1]

    float* out_feat = (float*)d_out;                       // [B,T,F]
    int*   out_hlen = (int*)d_out + OUT_FEAT_ELEMS;        // [B] int32 bits

    // Workspace layout: score [B*T] f32, srcIdx [B*T] i32
    float* score  = (float*)d_ws;
    int*   srcIdx = (int*)((char*)d_ws + (size_t)NFRAMES * sizeof(float));

    // Phase 1: 32768 frames / 16 per wave = 2048 waves; 8 waves per block.
    sigmax_score_wmma<<<NFRAMES / 16 / 8, 256, 0, stream>>>(feat, W, bias, score);

    // Phase 2: one block per batch.
    sigmax_peaks_scan<<<BDIM, 1024, 0, stream>>>(score, srcIdx, out_hlen);

    // Phase 3: one block per output row.
    sigmax_gather<<<NFRAMES, 128, 0, stream>>>(feat, srcIdx, out_hlen, out_feat);
}
